// MambaBlock_12094627906086
// MI455X (gfx1250) — compile-verified, matrix operand order checked by reading, not run on hardware
//
#include <hip/hip_runtime.h>
#include <hip/hip_bf16.h>

// Problem constants (fixed by the reference)
#define B_      4
#define L_      2048
#define DMODEL  1024
#define DSTATE  64
#define DCONV   4
#define DINNER  2048
#define DTRANK  64
#define NTOK    (B_ * L_)             // 8192 rows
#define XDBL_N  (DTRANK + 2 * DSTATE) // 192
#define EPSV    1e-6f

typedef __bf16 bf16_t;
typedef __attribute__((ext_vector_type(16))) __bf16 v16bf;
typedef __attribute__((ext_vector_type(8)))  float  v8f;

union FragBF { v16bf v; uint4 q[2]; };

// ---------------------------------------------------------------------------
// f32 -> bf16 bulk convert (weights)
// ---------------------------------------------------------------------------
__global__ __launch_bounds__(256) void f32_to_bf16_kernel(
    const float* __restrict__ in, __hip_bfloat16* __restrict__ out, size_t n) {
  size_t i = (size_t)blockIdx.x * 256 + threadIdx.x;
  if (i < n) out[i] = __float2bfloat16(in[i]);
}

// ---------------------------------------------------------------------------
// RMSNorm: one block per token row, 256 threads x 4 floats = 1024
// ---------------------------------------------------------------------------
__global__ __launch_bounds__(256) void rmsnorm_kernel(
    const float* __restrict__ x, const float* __restrict__ w,
    __hip_bfloat16* __restrict__ xn) {
  __shared__ float red[8];
  const size_t row = blockIdx.x;
  const float4 v = ((const float4*)(x + row * DMODEL))[threadIdx.x];
  float ss = v.x * v.x + v.y * v.y + v.z * v.z + v.w * v.w;
#pragma unroll
  for (int off = 16; off > 0; off >>= 1) ss += __shfl_down(ss, off, 32);
  if ((threadIdx.x & 31) == 0) red[threadIdx.x >> 5] = ss;
  __syncthreads();
  float tot = 0.f;
#pragma unroll
  for (int i = 0; i < 8; i++) tot += red[i];
  const float rs = rsqrtf(tot * (1.0f / DMODEL) + EPSV);
  const float4 wv = ((const float4*)w)[threadIdx.x];
  __hip_bfloat16* o = xn + row * DMODEL + threadIdx.x * 4;
  o[0] = __float2bfloat16(v.x * rs * wv.x);
  o[1] = __float2bfloat16(v.y * rs * wv.y);
  o[2] = __float2bfloat16(v.z * rs * wv.z);
  o[3] = __float2bfloat16(v.w * rs * wv.w);
}

// ---------------------------------------------------------------------------
// Large WMMA bf16 GEMM with async global->LDS double-buffered staging.
//   C[M,N] = A[M,K] * W[N,K]^T
// Block = 4 waves (2x2), tile 128x128, K-step 64 (2 WMMA sub-steps).
// LDS: 2 buffers x (A tile + B tile), rows padded to 144B for bank spread.
// Data movement uses GLOBAL_LOAD_ASYNC_TO_LDS_B128 (ASYNCcnt) so the async
// engine fills the next tile while WMMAs consume the current one.
// EPI: 1 = bf16 store, 3 = c + resid f32 store.
// ---------------------------------------------------------------------------
template <int EPI>
__global__ __launch_bounds__(128) void gemm_bf16_lds_kernel(
    const bf16_t* __restrict__ A, const bf16_t* __restrict__ W,
    float* __restrict__ Cf, __hip_bfloat16* __restrict__ Cb,
    const float* __restrict__ resid, int M, int N, int K) {
  constexpr int ROWB  = 144;          // padded LDS row stride (64 bf16 + 8 pad)
  constexpr int TILEB = 128 * ROWB;   // 18432 B per operand tile
  __shared__ __attribute__((aligned(16))) char smem[4 * TILEB];  // 72 KB

  const int tid    = threadIdx.x;
  const int lane   = tid & 31;
  const int wave   = tid >> 5;
  const int wm     = wave & 1;
  const int wn     = wave >> 1;
  const int mBlock = blockIdx.y * 128;
  const int nBlock = blockIdx.x * 128;
  const uint32_t ldsBase = (uint32_t)(uintptr_t)smem;  // low 32 bits = LDS offset

  // Issue one 128x64 A tile + 128x64 B tile: 16 async b128 loads per thread.
  auto issue = [&](int kt, int buf) {
    const int k0 = kt * 64;
    const uint32_t bufA = ldsBase + (uint32_t)buf * 2u * TILEB;
    const uint32_t bufB = bufA + TILEB;
#pragma unroll
    for (int it = 0; it < 8; ++it) {
      const int id  = it * 128 + tid;
      const int row = id >> 3;          // 0..127
      const int ch  = id & 7;           // 16B chunk within 128B row
      const uint64_t ga =
          (uint64_t)(uintptr_t)(A + (size_t)(mBlock + row) * K + k0 + ch * 8);
      const uint32_t la = bufA + (uint32_t)(row * ROWB + ch * 16);
      asm volatile("global_load_async_to_lds_b128 %0, %1, off"
                   :: "v"(la), "v"(ga) : "memory");
      const uint64_t gb =
          (uint64_t)(uintptr_t)(W + (size_t)(nBlock + row) * K + k0 + ch * 8);
      const uint32_t lb = bufB + (uint32_t)(row * ROWB + ch * 16);
      asm volatile("global_load_async_to_lds_b128 %0, %1, off"
                   :: "v"(lb), "v"(gb) : "memory");
    }
  };

  v8f acc[4][4] = {};
  const int rsel  = lane & 15;
  const int koffB = (lane >> 4) * 16;   // K sub-offset (8 bf16 = 16 B) per half-wave
  const int kIter = K / 64;

  issue(0, 0);
  for (int kt = 0; kt < kIter; ++kt) {
    if (kt + 1 < kIter) {
      issue(kt + 1, (kt + 1) & 1);
      asm volatile("s_wait_asynccnt 16" ::: "memory");  // tile kt landed; kt+1 in flight
    } else {
      asm volatile("s_wait_asynccnt 0" ::: "memory");
    }
    __syncthreads();  // all waves' async fills for tile kt visible

    const char* bufA = smem + (size_t)(kt & 1) * 2u * TILEB;
    const char* bufB = bufA + TILEB;
#pragma unroll
    for (int ks = 0; ks < 2; ++ks) {    // two 32-wide K sub-steps
      FragBF a[4], b[4];
#pragma unroll
      for (int i = 0; i < 4; i++) {
        const char* p = bufA + (wm * 64 + i * 16 + rsel) * ROWB + ks * 64 + koffB;
        a[i].q[0] = *(const uint4*)p;
        a[i].q[1] = *(const uint4*)(p + 32);
      }
#pragma unroll
      for (int j = 0; j < 4; j++) {
        const char* p = bufB + (wn * 64 + j * 16 + rsel) * ROWB + ks * 64 + koffB;
        b[j].q[0] = *(const uint4*)p;
        b[j].q[1] = *(const uint4*)(p + 32);
      }
#pragma unroll
      for (int i = 0; i < 4; i++)
#pragma unroll
        for (int j = 0; j < 4; j++)
          acc[i][j] = __builtin_amdgcn_wmma_f32_16x16x32_bf16(
              false, a[i].v, false, b[j].v, (short)0, acc[i][j], false, false);
    }
    __syncthreads();  // readers done before next issue overwrites this buffer
  }

  // C/D layout: VGPR r, lanes 0-15 -> M=r, lanes 16-31 -> M=r+8 ; N = lane&15
  const int mBase = mBlock + wm * 64;
  const int nBase = nBlock + wn * 64;
  const int cn    = lane & 15;
  const int rAdd  = (lane >> 4) * 8;
#pragma unroll
  for (int i = 0; i < 4; i++)
#pragma unroll
    for (int j = 0; j < 4; j++) {
      const int n = nBase + j * 16 + cn;
#pragma unroll
      for (int r = 0; r < 8; r++) {
        const int m = mBase + i * 16 + rAdd + r;
        const float v = acc[i][j][r];
        const size_t idx = (size_t)m * N + n;
        if (EPI == 1) Cb[idx] = __float2bfloat16(v);
        else          Cf[idx] = v + resid[idx];
      }
    }
}

// ---------------------------------------------------------------------------
// Direct WMMA bf16 GEMM (no LDS) for the narrow/shallow projections.
// Wave tile 64x64 (4x4 fragments), block = BM x BN waves.
// EPI: 0 = f32 store, 2 = softplus(c + bias[n]) f32 store.
// ---------------------------------------------------------------------------
template <int BM, int BN, int EPI>
__global__ __launch_bounds__(BM * BN * 32) void gemm_bf16_kernel(
    const bf16_t* __restrict__ A, const bf16_t* __restrict__ W,
    float* __restrict__ Cf, const float* __restrict__ bias,
    int M, int N, int K) {
  const int lane  = threadIdx.x & 31;
  const int wave  = threadIdx.x >> 5;
  const int wm    = wave % BM;
  const int wn    = wave / BM;
  const int mBase = (blockIdx.y * BM + wm) * 64;
  const int nBase = (blockIdx.x * BN + wn) * 64;
  const int rsel  = lane & 15;
  const int koff  = (lane >> 4) * 8;

  v8f acc[4][4] = {};

  for (int k0 = 0; k0 < K; k0 += 32) {
    FragBF a[4], b[4];
#pragma unroll
    for (int i = 0; i < 4; i++) {
      const bf16_t* p = A + (size_t)(mBase + i * 16 + rsel) * K + k0 + koff;
      a[i].q[0] = *(const uint4*)p;
      a[i].q[1] = *(const uint4*)(p + 16);
    }
#pragma unroll
    for (int j = 0; j < 4; j++) {
      const bf16_t* p = W + (size_t)(nBase + j * 16 + rsel) * K + k0 + koff;
      b[j].q[0] = *(const uint4*)p;
      b[j].q[1] = *(const uint4*)(p + 16);
    }
    if (k0 + 32 < K) {  // near-scope prefetch of next K tile
      __builtin_prefetch((const void*)(A + (size_t)(mBase + rsel) * K + k0 + 32), 0, 3);
      __builtin_prefetch((const void*)(W + (size_t)(nBase + rsel) * K + k0 + 32), 0, 3);
    }
#pragma unroll
    for (int i = 0; i < 4; i++)
#pragma unroll
      for (int j = 0; j < 4; j++)
        acc[i][j] = __builtin_amdgcn_wmma_f32_16x16x32_bf16(
            false, a[i].v, false, b[j].v, (short)0, acc[i][j], false, false);
  }

  const int cn   = lane & 15;
  const int rAdd = (lane >> 4) * 8;
#pragma unroll
  for (int i = 0; i < 4; i++)
#pragma unroll
    for (int j = 0; j < 4; j++) {
      const int n = nBase + j * 16 + cn;
#pragma unroll
      for (int r = 0; r < 8; r++) {
        const int m = mBase + i * 16 + rAdd + r;
        const float v = acc[i][j][r];
        const size_t idx = (size_t)m * N + n;
        if (EPI == 0) {
          Cf[idx] = v;
        } else {
          const float t = v + bias[n];
          Cf[idx] = (t > 20.f) ? t : log1pf(__expf(t));
        }
      }
    }
}

// ---------------------------------------------------------------------------
// Causal depthwise conv1d (k=4) + bias + SiLU.  Reads x_in half of xz (bf16).
// ---------------------------------------------------------------------------
__global__ __launch_bounds__(256) void conv_silu_kernel(
    const __hip_bfloat16* __restrict__ xz, const float* __restrict__ cw,
    const float* __restrict__ cb, __hip_bfloat16* __restrict__ u) {
  const size_t idx = (size_t)blockIdx.x * 256 + threadIdx.x;  // NTOK*DINNER
  const int c = (int)(idx % DINNER);
  const size_t bl = idx / DINNER;
  const int l = (int)(bl % L_);
  const size_t b = bl / L_;
  float acc = cb[c];
#pragma unroll
  for (int j = 0; j < DCONV; j++) {
    const int lt = l - (DCONV - 1) + j;
    if (lt >= 0) {
      const float xv = __bfloat162float(
          xz[((b * L_ + (size_t)lt) * (size_t)(2 * DINNER)) + c]);
      acc += cw[c * DCONV + j] * xv;
    }
  }
  u[idx] = __float2bfloat16(acc / (1.f + __expf(-acc)));  // SiLU
}

// ---------------------------------------------------------------------------
// Slice dt (first 64 cols of x_dbl) to bf16 for the dt_proj GEMM
// ---------------------------------------------------------------------------
__global__ __launch_bounds__(256) void slice_dt_kernel(
    const float* __restrict__ xdbl, __hip_bfloat16* __restrict__ dtb) {
  const size_t i = (size_t)blockIdx.x * 256 + threadIdx.x;  // NTOK*DTRANK
  const size_t r = i >> 6;
  const int c = (int)(i & 63);
  dtb[i] = __float2bfloat16(xdbl[r * XDBL_N + c]);
}

// ---------------------------------------------------------------------------
// Selective scan, fused with  y += u*D  and the SiLU(z) gate.
// block = 256 channels of one batch; per-lane state h[64] + A[64] in VGPRs,
// B_t / C_t broadcast through LDS each timestep.
// ---------------------------------------------------------------------------
__global__ __launch_bounds__(256) void scan_kernel(
    const float* __restrict__ xdbl, const float* __restrict__ delta,
    const __hip_bfloat16* __restrict__ u, const __hip_bfloat16* __restrict__ xz,
    const float* __restrict__ A_log, const float* __restrict__ Dv,
    __hip_bfloat16* __restrict__ yg) {
  __shared__ float bc[2 * DSTATE];
  const int d = blockIdx.x * 256 + threadIdx.x;  // channel
  const int b = blockIdx.y;                      // batch
  const float LOG2E = 1.44269504f;
  float An[DSTATE], h[DSTATE];
#pragma unroll
  for (int n = 0; n < DSTATE; n++) {
    An[n] = -__expf(A_log[(size_t)d * DSTATE + n]) * LOG2E;  // A * log2(e)
    h[n] = 0.f;
  }
  const float Dd = Dv[d];
  for (int t = 0; t < L_; t++) {
    const size_t row = (size_t)b * L_ + t;
    __syncthreads();
    if (threadIdx.x < 2 * DSTATE)
      bc[threadIdx.x] = xdbl[row * XDBL_N + DTRANK + threadIdx.x];
    __syncthreads();
    const float dt = delta[row * DINNER + d];
    const float ut = __bfloat162float(u[row * DINNER + d]);
    const float du = dt * ut;
    float y = 0.f;
#pragma unroll
    for (int n = 0; n < DSTATE; n++) {
      h[n] = exp2f(dt * An[n]) * h[n] + du * bc[n];  // dA*h + dB*u
      y += h[n] * bc[DSTATE + n];                    // h . C_t
    }
    const float yv = y + ut * Dd;
    const float z = __bfloat162float(xz[row * (size_t)(2 * DINNER) + DINNER + d]);
    yg[row * DINNER + d] = __float2bfloat16(yv * (z / (1.f + __expf(-z))));
  }
}

// ---------------------------------------------------------------------------
// Host-side orchestration
// ---------------------------------------------------------------------------
extern "C" void kernel_launch(void* const* d_in, const int* in_sizes, int n_in,
                              void* d_out, int out_size, void* d_ws, size_t ws_size,
                              hipStream_t stream) {
  (void)in_sizes; (void)n_in; (void)out_size; (void)ws_size;
  const float* x         = (const float*)d_in[0];
  const float* norm_w    = (const float*)d_in[1];
  const float* in_proj_w = (const float*)d_in[2];
  const float* conv_w    = (const float*)d_in[3];
  const float* conv_b    = (const float*)d_in[4];
  const float* x_proj_w  = (const float*)d_in[5];
  const float* dt_proj_w = (const float*)d_in[6];
  const float* dt_proj_b = (const float*)d_in[7];
  const float* A_log     = (const float*)d_in[8];
  const float* Dvec      = (const float*)d_in[9];
  const float* out_proj_w= (const float*)d_in[10];
  float* out = (float*)d_out;

  char* wp = (char*)d_ws;
  auto alloc = [&](size_t bytes) -> void* {
    void* p = (void*)wp;
    wp += (bytes + 255) & ~(size_t)255;
    return p;
  };
  __hip_bfloat16* wIn  = (__hip_bfloat16*)alloc((size_t)2 * DINNER * DMODEL * 2);  // [4096,1024]
  __hip_bfloat16* wXp  = (__hip_bfloat16*)alloc((size_t)XDBL_N * DINNER * 2);      // [192,2048]
  __hip_bfloat16* wDt  = (__hip_bfloat16*)alloc((size_t)DINNER * DTRANK * 2);      // [2048,64]
  __hip_bfloat16* wOut = (__hip_bfloat16*)alloc((size_t)DMODEL * DINNER * 2);      // [1024,2048]
  __hip_bfloat16* xn   = (__hip_bfloat16*)alloc((size_t)NTOK * DMODEL * 2);        // [8192,1024]
  __hip_bfloat16* xz   = (__hip_bfloat16*)alloc((size_t)NTOK * 2 * DINNER * 2);    // [8192,4096]
  __hip_bfloat16* ub   = (__hip_bfloat16*)alloc((size_t)NTOK * DINNER * 2);        // [8192,2048]
  float*          xdbl = (float*)alloc((size_t)NTOK * XDBL_N * 4);                 // [8192,192]
  __hip_bfloat16* dtb  = (__hip_bfloat16*)alloc((size_t)NTOK * DTRANK * 2);        // [8192,64]
  float*          dlt  = (float*)alloc((size_t)NTOK * DINNER * 4);                 // [8192,2048]
  __hip_bfloat16* yg   = (__hip_bfloat16*)alloc((size_t)NTOK * DINNER * 2);        // [8192,2048]

  auto cvt = [&](const float* src, __hip_bfloat16* dst, size_t n) {
    f32_to_bf16_kernel<<<(unsigned)((n + 255) / 256), 256, 0, stream>>>(src, dst, n);
  };
  cvt(in_proj_w, wIn, (size_t)2 * DINNER * DMODEL);
  cvt(x_proj_w,  wXp, (size_t)XDBL_N * DINNER);
  cvt(dt_proj_w, wDt, (size_t)DINNER * DTRANK);
  cvt(out_proj_w,wOut,(size_t)DMODEL * DINNER);

  // 1) RMSNorm -> xn (bf16)
  rmsnorm_kernel<<<NTOK, 256, 0, stream>>>(x, norm_w, xn);

  // 2) in_proj: xz = xn * in_proj_w^T  (8192 x 4096 x 1024), async-LDS GEMM, bf16 out
  gemm_bf16_lds_kernel<1><<<dim3(4096 / 128, 8192 / 128), 128, 0, stream>>>(
      (const bf16_t*)xn, (const bf16_t*)wIn, nullptr, xz, nullptr,
      NTOK, 2 * DINNER, DMODEL);

  // 3) causal depthwise conv + SiLU -> u (bf16)
  conv_silu_kernel<<<(unsigned)(((size_t)NTOK * DINNER) / 256), 256, 0, stream>>>(
      xz, conv_w, conv_b, ub);

  // 4) x_proj: x_dbl = u * x_proj_w^T  (8192 x 192 x 2048), f32 out
  gemm_bf16_kernel<4, 1, 0><<<dim3(XDBL_N / 64, 8192 / 256), 128, 0, stream>>>(
      (const bf16_t*)ub, (const bf16_t*)wXp, xdbl, nullptr,
      NTOK, XDBL_N, DINNER);

  // 5) dt slice -> bf16
  slice_dt_kernel<<<(unsigned)(((size_t)NTOK * DTRANK) / 256), 256, 0, stream>>>(xdbl, dtb);

  // 6) dt_proj + softplus: delta = softplus(dt * dt_proj_w^T + b)  (8192 x 2048 x 64)
  gemm_bf16_kernel<2, 2, 2><<<dim3(2048 / 128, 8192 / 128), 128, 0, stream>>>(
      (const bf16_t*)dtb, (const bf16_t*)wDt, dlt, dt_proj_b,
      NTOK, DINNER, DTRANK);

  // 7) selective scan fused with +u*D and SiLU(z) gate -> yg (bf16)
  scan_kernel<<<dim3(DINNER / 256, B_), 256, 0, stream>>>(
      xdbl, dlt, ub, xz, A_log, Dvec, yg);

  // 8) out_proj + residual: out = yg * out_proj_w^T + x  (8192 x 1024 x 2048), async-LDS GEMM
  gemm_bf16_lds_kernel<3><<<dim3(1024 / 128, 8192 / 128), 128, 0, stream>>>(
      (const bf16_t*)yg, (const bf16_t*)wOut, out, nullptr, x,
      NTOK, DMODEL, DINNER);
}
